// SequentialAudioVisualModel_80461917323525
// MI455X (gfx1250) — matrix operand order repaired and checked by reading, not compiled, not run on hardware
//
#include <hip/hip_runtime.h>

#define B_    32
#define T_    40
#define P_    256
#define V_    4096
#define D_    256
#define DINO_ 384
#define H_    4
#define L_    2
#define HD_   64
#define DFF_  1024
#define MBT   (B_ * T_)
#define MBP   (B_ * P_)
#define PBLK  32

typedef __attribute__((ext_vector_type(16))) __bf16 v16bf;
typedef __attribute__((ext_vector_type(8)))  __bf16 v8bf;
typedef __attribute__((ext_vector_type(8)))  float  v8f;

// --------------------------------------------------------------------------
// gfx1250 async global->LDS copies (ASYNCcnt path), with portable fallback.
// Builtin signatures (from hipcc diagnostics):
//   b32 : (AS1 int*  g, AS3 int*  lds, int offset, int cpol)
//   b128: (AS1 int4* g, AS3 int4* lds, int offset, int cpol)
// --------------------------------------------------------------------------
#if defined(__HIP_DEVICE_COMPILE__) && defined(__gfx1250__) && \
    __has_builtin(__builtin_amdgcn_global_load_async_to_lds_b128) && \
    __has_builtin(__builtin_amdgcn_global_load_async_to_lds_b32)
#define HAS_ASYNC 1
#else
#define HAS_ASYNC 0
#endif

#define AS1_ __attribute__((address_space(1)))
#define AS3_ __attribute__((address_space(3)))
typedef __attribute__((ext_vector_type(4))) int i4v;

__device__ __forceinline__ void async_copy_b128(void* lds, const void* g) {
#if HAS_ASYNC
  __builtin_amdgcn_global_load_async_to_lds_b128(
      (AS1_ i4v*)(AS1_ void*)g,
      (AS3_ i4v*)(AS3_ void*)lds, 0, 0);
#else
  *(float4*)lds = *(const float4*)g;
#endif
}

__device__ __forceinline__ void async_copy_b32(void* lds, const void* g) {
#if HAS_ASYNC
  __builtin_amdgcn_global_load_async_to_lds_b32(
      (AS1_ int*)(AS1_ void*)g,
      (AS3_ int*)(AS3_ void*)lds, 0, 0);
#else
  *(float*)lds = *(const float*)g;
#endif
}

__device__ __forceinline__ void wait_async_all() {
#if HAS_ASYNC
#if __has_builtin(__builtin_amdgcn_s_wait_asynccnt)
  __builtin_amdgcn_s_wait_asynccnt(0);
#else
  asm volatile("s_wait_asynccnt 0" ::: "memory");
#endif
#endif
}

__device__ __forceinline__ unsigned short f2bf(float f) {
  union { float f; unsigned u; } v; v.f = f;
  unsigned r = v.u + 0x7FFFu + ((v.u >> 16) & 1u);   // round-to-nearest-even
  return (unsigned short)(r >> 16);
}

// ---------------------------------------------------------------------------
// Conversion kernels (f32 -> bf16 stored as u16)
// ---------------------------------------------------------------------------
__global__ void convert_kernel(const float* __restrict__ src,
                               unsigned short* __restrict__ dst, int n) {
  int i = blockIdx.x * blockDim.x + threadIdx.x;
  if (i < n) dst[i] = f2bf(src[i]);
}

__global__ void convert_slice_kernel(const float* __restrict__ src,
                                     unsigned short* __restrict__ dst,
                                     int rows, int cols, int srcStride, int srcOff) {
  int i = blockIdx.x * blockDim.x + threadIdx.x;
  if (i >= rows * cols) return;
  int r = i / cols, c = i % cols;
  dst[i] = f2bf(src[(size_t)r * srcStride + srcOff + c]);
}

__global__ void transpose_bf16_kernel(const float* __restrict__ src,
                                      unsigned short* __restrict__ dst,
                                      int rows, int cols) {
  int i = blockIdx.x * blockDim.x + threadIdx.x;
  if (i >= rows * cols) return;
  int r = i / cols, c = i % cols;
  dst[(size_t)c * rows + r] = f2bf(src[i]);
}

// ---------------------------------------------------------------------------
// Embedding: x = tok_emb[tokens] + pos_emb  (emit f32 + bf16)
// ---------------------------------------------------------------------------
__global__ void embed_kernel(const int* __restrict__ tokens,
                             const float* __restrict__ tok_emb,
                             const float* __restrict__ pos_emb,
                             float* __restrict__ xf,
                             unsigned short* __restrict__ xb) {
  int i = blockIdx.x * 256 + threadIdx.x;       // grid = MBT blocks of 256
  int d = i & (D_ - 1);
  int m = i >> 8;                               // D_ == 256
  int t = m % T_;
  float v = tok_emb[(size_t)tokens[m] * D_ + d] + pos_emb[(size_t)t * D_ + d];
  xf[i] = v;
  xb[i] = f2bf(v);
}

// ---------------------------------------------------------------------------
// bf16 WMMA GEMM:  C[M,N] = act(A[M,K] @ B[N,K]^T + bias + resid)
// 16x64 tile per wave: one A fragment re-used across 4 B fragments / WMMAs.
// ---------------------------------------------------------------------------
__global__ void gemm_bf16_kernel(const unsigned short* __restrict__ A,
                                 const unsigned short* __restrict__ Bw,
                                 const float* __restrict__ bias,
                                 const float* __restrict__ resid,
                                 float* __restrict__ Cf,
                                 unsigned short* __restrict__ Cb,
                                 int M, int N, int K, int relu) {
  const int wid  = blockIdx.x * (blockDim.x >> 5) + (threadIdx.x >> 5);
  const int lane = threadIdx.x & 31;
  const int hf   = lane >> 4;
  const int l16  = lane & 15;
  const int tn   = N >> 6;                      // 64-wide N tiles
  if (wid >= (M >> 4) * tn) return;             // wave-uniform guard
  const int m0 = (wid / tn) << 4;
  const int n0 = (wid % tn) << 6;

  v8f acc0 = {}, acc1 = {}, acc2 = {}, acc3 = {};
  const unsigned short* aRow = A  + (size_t)(m0 + l16) * K + hf * 16;
  const unsigned short* bRow = Bw + (size_t)(n0 + l16) * K + hf * 16;
  const size_t bstep = (size_t)16 * K;

  for (int k0 = 0; k0 < K; k0 += 32) {
    v8bf  alo = *(const v8bf*)(const void*)(aRow + k0);
    v8bf  ahi = *(const v8bf*)(const void*)(aRow + k0 + 8);
    v16bf av;
#pragma unroll
    for (int i = 0; i < 8; ++i) { av[i] = alo[i]; av[i + 8] = ahi[i]; }
    v16bf bv0 = *(const v16bf*)(const void*)(bRow + k0);
    v16bf bv1 = *(const v16bf*)(const void*)(bRow + bstep + k0);
    v16bf bv2 = *(const v16bf*)(const void*)(bRow + 2 * bstep + k0);
    v16bf bv3 = *(const v16bf*)(const void*)(bRow + 3 * bstep + k0);
    acc0 = __builtin_amdgcn_wmma_f32_16x16x32_bf16(false, av, false, bv0,
                                                   (short)0, acc0, false, false);
    acc1 = __builtin_amdgcn_wmma_f32_16x16x32_bf16(false, av, false, bv1,
                                                   (short)0, acc1, false, false);
    acc2 = __builtin_amdgcn_wmma_f32_16x16x32_bf16(false, av, false, bv2,
                                                   (short)0, acc2, false, false);
    acc3 = __builtin_amdgcn_wmma_f32_16x16x32_bf16(false, av, false, bv3,
                                                   (short)0, acc3, false, false);
  }

#pragma unroll
  for (int j = 0; j < 4; ++j) {
    v8f acc = (j == 0) ? acc0 : (j == 1) ? acc1 : (j == 2) ? acc2 : acc3;
    const int col = n0 + j * 16 + l16;
    const float bcol = bias ? bias[col] : 0.0f;
#pragma unroll
    for (int r = 0; r < 8; ++r) {
      int row = m0 + hf * 8 + r;
      float v = acc[r] + bcol;
      if (resid) v += resid[(size_t)row * N + col];
      if (relu)  v = v > 0.0f ? v : 0.0f;
      size_t idx = (size_t)row * N + col;
      if (Cf) Cf[idx] = v;
      if (Cb) Cb[idx] = f2bf(v);
    }
  }
}

// ---------------------------------------------------------------------------
// Causal attention, one (b,h) per block (T=40, HD=64). Emits bf16 ctx.
// ---------------------------------------------------------------------------
__global__ void attn_kernel(const float* __restrict__ qkv,
                            unsigned short* __restrict__ ctx_bf) {
  __shared__ float qs[T_][HD_ + 1];
  __shared__ float ks[T_][HD_ + 1];
  __shared__ float vs[T_][HD_ + 1];
  __shared__ float sc[T_][T_ + 1];

  const int bh = blockIdx.x;
  const int b = bh / H_, h = bh % H_;
  const float* base = qkv + (size_t)b * T_ * 3 * D_ + h * HD_;

  for (int idx = threadIdx.x; idx < T_ * HD_; idx += blockDim.x) {
    int t = idx / HD_, d = idx % HD_;
    const float* row = base + (size_t)t * 3 * D_ + d;
    qs[t][d] = row[0];
    ks[t][d] = row[D_];
    vs[t][d] = row[2 * D_];
  }
  __syncthreads();

  const int qt = threadIdx.x;
  if (qt < T_) {
    float mx = -1e30f;
    for (int j = 0; j <= qt; ++j) {
      float s = 0.0f;
#pragma unroll 8
      for (int d = 0; d < HD_; ++d) s += qs[qt][d] * ks[j][d];
      s *= 0.125f;                      // 1/sqrt(HD)
      sc[qt][j] = s;
      mx = fmaxf(mx, s);
    }
    float sum = 0.0f;
    for (int j = 0; j <= qt; ++j) {
      float e = __expf(sc[qt][j] - mx);
      sc[qt][j] = e;
      sum += e;
    }
    const float inv = 1.0f / sum;
    unsigned short* outp = ctx_bf + ((size_t)(b * T_ + qt)) * D_ + h * HD_;
    for (int d = 0; d < HD_; ++d) {
      float acc = 0.0f;
      for (int j = 0; j <= qt; ++j) acc += sc[qt][j] * vs[j][d];
      outp[d] = f2bf(acc * inv);
    }
  }
}

// ---------------------------------------------------------------------------
// Row LayerNorm over D=256, one row per block of 256 threads. Emits f32+bf16.
// ---------------------------------------------------------------------------
__global__ void ln_kernel(const float* __restrict__ y,
                          const float* __restrict__ g,
                          const float* __restrict__ bta,
                          float* __restrict__ xf,
                          unsigned short* __restrict__ xb) {
  __shared__ float rs[256], rq[256];
  const int m = blockIdx.x, tid = threadIdx.x;
  float v = y[(size_t)m * D_ + tid];
  rs[tid] = v; rq[tid] = v * v;
  __syncthreads();
  for (int off = 128; off > 0; off >>= 1) {
    if (tid < off) { rs[tid] += rs[tid + off]; rq[tid] += rq[tid + off]; }
    __syncthreads();
  }
  float mean = rs[0] * (1.0f / D_);
  float var  = rq[0] * (1.0f / D_) - mean * mean;
  float inv  = rsqrtf(var + 1e-5f);
  float o = (v - mean) * inv * g[tid] + bta[tid];
  xf[(size_t)m * D_ + tid] = o;
  xb[(size_t)m * D_ + tid] = f2bf(o);
}

// ---------------------------------------------------------------------------
// Fusion prep, one (b,t)=m per block of 256 threads.
// ---------------------------------------------------------------------------
__global__ void fusion_prep_kernel(const int* __restrict__ tokens,
                                   const float* __restrict__ cls_w,
                                   const float* __restrict__ cls_b,
                                   const float* __restrict__ fus_b2,
                                   const unsigned short* __restrict__ enc_bf,
                                   unsigned short* __restrict__ wsel_bf,
                                   unsigned short* __restrict__ ctxf_bf,
                                   float* __restrict__ S) {
  __shared__ float rs[256];
  const int m = blockIdx.x, tid = threadIdx.x;
  const int t = m % T_;
  const int tok = tokens[m];
  float w = cls_w[(size_t)tok * D_ + tid];
  wsel_bf[(size_t)m * D_ + tid] = f2bf(w);
  ctxf_bf[(size_t)m * D_ + tid] = (t == 0) ? (unsigned short)0
                                           : enc_bf[(size_t)(m - 1) * D_ + tid];
  rs[tid] = fus_b2[tid] * w;
  __syncthreads();
  for (int off = 128; off > 0; off >>= 1) {
    if (tid < off) rs[tid] += rs[tid + off];
    __syncthreads();
  }
  if (tid == 0) S[m] = rs[0] + cls_b[tok];
}

// ---------------------------------------------------------------------------
// Final fused kernel, one (b, patch-block of 32) per block:
//   async-DMA the pproj tile into LDS (transposed, conflict-free), reuse over
//   all T timesteps; per-t cproj/u vectors staged via async b128:
//   logits[b,t,p] = sum_d gelu(pproj[b,p,d] + cproj[b,t,d]) * u[b,t,d] + S[b,t]
//   out[b,p,t]    = sigmoid(logits)
// ---------------------------------------------------------------------------
__global__ void final_kernel(const float* __restrict__ pproj,
                             const float* __restrict__ cproj,
                             const float* __restrict__ u,
                             const float* __restrict__ S,
                             float* __restrict__ outp) {
  __shared__ float ptile_t[D_][PBLK + 1];            // [d][p], stride 33
  __shared__ __align__(16) float cs[D_];
  __shared__ __align__(16) float us[D_];
  __shared__ float red[8][PBLK];

  const int blk = blockIdx.x;
  const int b   = blk / (P_ / PBLK);
  const int p0  = (blk % (P_ / PBLK)) * PBLK;
  const int tid = threadIdx.x;

  // Async global->LDS: coalesced global reads, stride-33 LDS writes.
  const float* pb = pproj + ((size_t)b * P_ + p0) * D_;
  for (int idx = tid; idx < PBLK * D_; idx += 256) {
    int pl = idx / D_, d = idx % D_;
    async_copy_b32(&ptile_t[d][pl], pb + idx);
  }
  wait_async_all();
  __syncthreads();

  const int pl = tid & (PBLK - 1);
  const int dq = tid >> 5;                    // 0..7, 32 dims each

  for (int t = 0; t < T_; ++t) {
    const int m = b * T_ + t;
    if (tid < 64) {
      async_copy_b128(&cs[tid * 4], cproj + (size_t)m * D_ + tid * 4);
    } else if (tid < 128) {
      int q = (tid - 64) * 4;
      async_copy_b128(&us[q], u + (size_t)m * D_ + q);
    }
    wait_async_all();
    __syncthreads();

    float acc = 0.0f;
    const int d0 = dq * 32;
#pragma unroll 8
    for (int d = d0; d < d0 + 32; ++d) {
      float xg = ptile_t[d][pl] + cs[d];
      float gl = 0.5f * xg * (1.0f + erff(xg * 0.70710678f));   // exact gelu
      acc += gl * us[d];
    }
    red[dq][pl] = acc;
    __syncthreads();

    if (tid < PBLK) {
      float lg = S[m];
#pragma unroll
      for (int j = 0; j < 8; ++j) lg += red[j][tid];
      outp[((size_t)b * P_ + p0 + tid) * T_ + t] = 1.0f / (1.0f + __expf(-lg));
    }
    __syncthreads();
  }
}

// ---------------------------------------------------------------------------
// Host orchestration
// ---------------------------------------------------------------------------
extern "C" void kernel_launch(void* const* d_in, const int* in_sizes, int n_in,
                              void* d_out, int out_size, void* d_ws, size_t ws_size,
                              hipStream_t stream) {
  (void)in_sizes; (void)n_in; (void)out_size; (void)ws_size;

  const float* patches = (const float*)d_in[0];
  const int*   tokens  = (const int*)d_in[1];
  const float* tok_emb = (const float*)d_in[2];
  const float* pos_emb = (const float*)d_in[3];
  const float* qkv_w   = (const float*)d_in[4];
  const float* qkv_b   = (const float*)d_in[5];
  const float* out_w   = (const float*)d_in[6];
  const float* out_b   = (const float*)d_in[7];
  const float* ln1_s   = (const float*)d_in[8];
  const float* ln1_b   = (const float*)d_in[9];
  const float* w1      = (const float*)d_in[10];
  const float* b1      = (const float*)d_in[11];
  const float* w2      = (const float*)d_in[12];
  const float* b2      = (const float*)d_in[13];
  const float* ln2_s   = (const float*)d_in[14];
  const float* ln2_b   = (const float*)d_in[15];
  const float* fus_w1  = (const float*)d_in[16];
  const float* fus_b1  = (const float*)d_in[17];
  const float* fus_w2  = (const float*)d_in[18];
  const float* fus_b2  = (const float*)d_in[19];
  const float* cls_w   = (const float*)d_in[20];
  const float* cls_b   = (const float*)d_in[21];
  float* out = (float*)d_out;

  char* ws = (char*)d_ws;
  size_t cur = 0;
  auto alloc = [&](size_t bytes) -> char* {
    char* p = ws + cur;
    cur += (bytes + 255) & ~(size_t)255;
    return p;
  };

  unsigned short* qkvw_bf = (unsigned short*)alloc((size_t)L_ * 3 * D_ * D_ * 2);
  unsigned short* outw_bf = (unsigned short*)alloc((size_t)L_ * D_ * D_ * 2);
  unsigned short* w1_bf   = (unsigned short*)alloc((size_t)L_ * DFF_ * D_ * 2);
  unsigned short* w2_bf   = (unsigned short*)alloc((size_t)L_ * D_ * DFF_ * 2);
  unsigned short* wp_bf   = (unsigned short*)alloc((size_t)D_ * DINO_ * 2);
  unsigned short* wc_bf   = (unsigned short*)alloc((size_t)D_ * D_ * 2);
  unsigned short* w2t_bf  = (unsigned short*)alloc((size_t)D_ * D_ * 2);
  unsigned short* patch_bf= (unsigned short*)alloc((size_t)B_ * P_ * DINO_ * 2);
  float*          xf      = (float*)alloc((size_t)MBT * D_ * 4);
  unsigned short* xb      = (unsigned short*)alloc((size_t)MBT * D_ * 2);
  float*          qkvf    = (float*)alloc((size_t)MBT * 3 * D_ * 4);
  unsigned short* ctxb    = (unsigned short*)alloc((size_t)MBT * D_ * 2);
  float*          yf      = (float*)alloc((size_t)MBT * D_ * 4);
  unsigned short* hb      = (unsigned short*)alloc((size_t)MBT * DFF_ * 2);
  float*          pproj   = (float*)alloc((size_t)MBP * D_ * 4);
  float*          cproj   = (float*)alloc((size_t)MBT * D_ * 4);
  float*          uu      = (float*)alloc((size_t)MBT * D_ * 4);
  unsigned short* wselb   = (unsigned short*)alloc((size_t)MBT * D_ * 2);
  unsigned short* ctxfb   = (unsigned short*)alloc((size_t)MBT * D_ * 2);
  float*          Sv      = (float*)alloc((size_t)MBT * 4);

  auto cvt = [&](const float* s, unsigned short* d, int n) {
    convert_kernel<<<(n + 255) / 256, 256, 0, stream>>>(s, d, n);
  };
  auto gemm = [&](const unsigned short* A, const unsigned short* Bw,
                  const float* bias, const float* resid,
                  float* Cf, unsigned short* Cb, int M, int N, int K, int relu) {
    int tiles = (M / 16) * (N / 64);
    gemm_bf16_kernel<<<(tiles + 7) / 8, 256, 0, stream>>>(A, Bw, bias, resid,
                                                          Cf, Cb, M, N, K, relu);
  };

  // Weight / input conversions (bf16, once per launch)
  cvt(qkv_w, qkvw_bf, L_ * 3 * D_ * D_);
  cvt(out_w, outw_bf, L_ * D_ * D_);
  cvt(w1, w1_bf, L_ * DFF_ * D_);
  cvt(w2, w2_bf, L_ * D_ * DFF_);
  cvt(patches, patch_bf, B_ * P_ * DINO_);
  convert_slice_kernel<<<(D_ * DINO_ + 255) / 256, 256, 0, stream>>>(
      fus_w1, wp_bf, D_, DINO_, DINO_ + D_, 0);
  convert_slice_kernel<<<(D_ * D_ + 255) / 256, 256, 0, stream>>>(
      fus_w1, wc_bf, D_, D_, DINO_ + D_, DINO_);
  transpose_bf16_kernel<<<(D_ * D_ + 255) / 256, 256, 0, stream>>>(
      fus_w2, w2t_bf, D_, D_);

  // Token encoder
  embed_kernel<<<MBT, 256, 0, stream>>>(tokens, tok_emb, pos_emb, xf, xb);
  for (int l = 0; l < L_; ++l) {
    gemm(xb, qkvw_bf + (size_t)l * 3 * D_ * D_, qkv_b + (size_t)l * 3 * D_,
         nullptr, qkvf, nullptr, MBT, 3 * D_, D_, 0);
    attn_kernel<<<B_ * H_, 64, 0, stream>>>(qkvf, ctxb);
    gemm(ctxb, outw_bf + (size_t)l * D_ * D_, out_b + (size_t)l * D_,
         xf, yf, nullptr, MBT, D_, D_, 0);
    ln_kernel<<<MBT, 256, 0, stream>>>(yf, ln1_s + (size_t)l * D_,
                                       ln1_b + (size_t)l * D_, xf, xb);
    gemm(xb, w1_bf + (size_t)l * DFF_ * D_, b1 + (size_t)l * DFF_,
         nullptr, nullptr, hb, MBT, DFF_, D_, 1);
    gemm(hb, w2_bf + (size_t)l * D_ * DFF_, b2 + (size_t)l * D_,
         xf, yf, nullptr, MBT, D_, DFF_, 0);
    ln_kernel<<<MBT, 256, 0, stream>>>(yf, ln2_s + (size_t)l * D_,
                                       ln2_b + (size_t)l * D_, xf, xb);
  }

  // Fusion stage (fus_w2 algebraically folded into per-token u vectors)
  fusion_prep_kernel<<<MBT, 256, 0, stream>>>(tokens, cls_w, cls_b, fus_b2,
                                              xb, wselb, ctxfb, Sv);
  gemm(patch_bf, wp_bf, nullptr, nullptr, pproj, nullptr, MBP, D_, DINO_, 0);
  gemm(ctxfb, wc_bf, fus_b1, nullptr, cproj, nullptr, MBT, D_, D_, 0);
  gemm(wselb, w2t_bf, nullptr, nullptr, uu, nullptr, MBT, D_, D_, 0);

  final_kernel<<<B_ * (P_ / PBLK), 256, 0, stream>>>(pproj, cproj, uu, Sv, out);
}
